// IRNAttentionTrans_395136991510
// MI455X (gfx1250) — matrix operand order repaired and checked by reading, not compiled, not run on hardware
//
#include <hip/hip_runtime.h>
#include <hip/hip_bf16.h>

#define B_ 64
#define N_ 1024
#define D_ 256
#define P_ 256

typedef __attribute__((ext_vector_type(16))) __bf16 v16bf;
typedef __attribute__((ext_vector_type(8)))  __bf16 v8bf;
typedef __attribute__((ext_vector_type(8)))  float  v8f;

union FragU { v16bf v; v8bf h[2]; };

// A-matrix fragment (16x32 bf16): lane L<16 -> row L, contiguous K..K+15;
// lanes 16..31 -> same rows, K+16..K+31 (matches B fragment k order).
__device__ __forceinline__ v16bf load_a_frag(const __bf16* base, int ld, int lane, int kbase) {
  const __bf16* p = base + (size_t)(lane & 15) * ld + kbase + ((lane >> 4) & 1) * 16;
  FragU f;
  f.h[0] = *(const v8bf*)(p);
  f.h[1] = *(const v8bf*)(p + 8);
  return f.v;
}

// B-matrix fragment (32x16 bf16), loaded from B^T row-major: lane L<16 -> col L,
// contiguous K..K+15; lanes 16..31 -> K+16..K+31 (matches A fragment k order).
__device__ __forceinline__ v16bf load_b_frag(const __bf16* base, int ld, int lane, int kbase) {
  const __bf16* p = base + (size_t)(lane & 15) * ld + kbase + ((lane >> 4) & 1) * 16;
  return *(const v16bf*)(p);
}

__device__ __forceinline__ float sigmoidf_fast(float x) {
  return 1.0f / (1.0f + __expf(-x));
}

// ---------------------------------------------------------------------------
// Kernel 1: x (f32, [B,N,D]) -> xbf (bf16 row-major) and xT (bf16, [B,D,N])
// ---------------------------------------------------------------------------
__global__ void k_prep_x(const float* __restrict__ x,
                         __bf16* __restrict__ xbf, __bf16* __restrict__ xT) {
  __shared__ float t[16][17];
  int b  = blockIdx.z;
  int n0 = blockIdx.y * 16;
  int d0 = blockIdx.x * 16;
  int tx = threadIdx.x, ty = threadIdx.y;
  float v = x[((size_t)b * N_ + n0 + ty) * D_ + d0 + tx];
  xbf[((size_t)b * N_ + n0 + ty) * D_ + d0 + tx] = (__bf16)v;
  t[ty][tx] = v;
  __syncthreads();
  xT[((size_t)b * D_ + d0 + ty) * N_ + n0 + tx] = (__bf16)t[tx][ty];
}

// ---------------------------------------------------------------------------
// Kernel 2: w1/w2 ([D,P] f32) -> w1T/w2T ([P,D] bf16)
// ---------------------------------------------------------------------------
__global__ void k_prep_w(const float* __restrict__ w1, const float* __restrict__ w2,
                         __bf16* __restrict__ w1T, __bf16* __restrict__ w2T) {
  __shared__ float t[16][17];
  const float* w = blockIdx.z ? w2 : w1;
  __bf16* wT = blockIdx.z ? w2T : w1T;
  int k0 = blockIdx.y * 16, p0 = blockIdx.x * 16;
  int tx = threadIdx.x, ty = threadIdx.y;
  t[ty][tx] = w[(size_t)(k0 + ty) * P_ + p0 + tx];
  __syncthreads();
  wT[(size_t)(p0 + ty) * D_ + k0 + tx] = (__bf16)t[tx][ty];
}

// ---------------------------------------------------------------------------
// Kernel 3: q = sigmoid(x@w1), k = sigmoid(x@w2), bf16 outputs.
// 8 waves/block; each wave owns a 16-row tile (shared A frags) and sweeps all
// 16 column tiles for BOTH weight matrices (2 WMMAs per K-step).
// ---------------------------------------------------------------------------
__global__ void __launch_bounds__(256) k_qk(const __bf16* __restrict__ xbf,
                                            const __bf16* __restrict__ w1T,
                                            const __bf16* __restrict__ w2T,
                                            __bf16* __restrict__ q,
                                            __bf16* __restrict__ kmat) {
  int wave = threadIdx.x >> 5;
  int lane = threadIdx.x & 31;
  size_t row0 = ((size_t)blockIdx.x * 8 + wave) * 16;   // over B*N rows
  const __bf16* arow = xbf + row0 * D_;

  v16bf af[8];
#pragma unroll
  for (int ks = 0; ks < 8; ++ks)
    af[ks] = load_a_frag(arow, D_, lane, ks * 32);

  int mrow = ((lane >> 4) & 1) * 8;
  int ncol = lane & 15;
  for (int ct = 0; ct < 16; ++ct) {
    v8f accq = {0.f, 0.f, 0.f, 0.f, 0.f, 0.f, 0.f, 0.f};
    v8f acck = accq;
    const __bf16* b1 = w1T + (size_t)ct * 16 * D_;
    const __bf16* b2 = w2T + (size_t)ct * 16 * D_;
#pragma unroll
    for (int ks = 0; ks < 8; ++ks) {
      v16bf bq = load_b_frag(b1, D_, lane, ks * 32);
      v16bf bk = load_b_frag(b2, D_, lane, ks * 32);
      accq = __builtin_amdgcn_wmma_f32_16x16x32_bf16(false, af[ks], false, bq,
                                                     (short)0, accq, false, false);
      acck = __builtin_amdgcn_wmma_f32_16x16x32_bf16(false, af[ks], false, bk,
                                                     (short)0, acck, false, false);
    }
#pragma unroll
    for (int j = 0; j < 8; ++j) {
      size_t r = row0 + mrow + j;
      q[r * P_ + ct * 16 + ncol]    = (__bf16)sigmoidf_fast(accq[j]);
      kmat[r * P_ + ct * 16 + ncol] = (__bf16)sigmoidf_fast(acck[j]);
    }
  }
}

// ---------------------------------------------------------------------------
// Kernel 4: fused attention row-tile. One block = (batch b, 16 query rows).
//  A) S[16][1024] = Q_tile @ K^T  (WMMA, f32 in LDS)
//  B) row softmax (scale 1/16) -> bf16 P in LDS
//  C) out_tile[16][256] = P @ x   (WMMA via xT; one A-frag feeds 2 accums)
//  D) logits[row] = out_tile @ w4 (the only consumer of `out`)
// ---------------------------------------------------------------------------
#define SPAD 1032
#define PPAD 1040
#define OPAD 264

__global__ void __launch_bounds__(256) k_attn(const __bf16* __restrict__ q,
                                              const __bf16* __restrict__ kmat,
                                              const __bf16* __restrict__ xT,
                                              const float*  __restrict__ w4,
                                              float* __restrict__ logits) {
  __shared__ float  S[16][SPAD];
  __shared__ __bf16 Pm[16][PPAD];
  __shared__ float  outb[16][OPAD];

  int wave = threadIdx.x >> 5;
  int lane = threadIdx.x & 31;
  int rt = blockIdx.x;
  int b  = blockIdx.y;
  int row0 = rt * 16;
  int mrow = ((lane >> 4) & 1) * 8;
  int ncol = lane & 15;

  // ---- Phase A: scores ----
  const __bf16* qrow = q + ((size_t)b * N_ + row0) * P_;
  v16bf qa[8];
#pragma unroll
  for (int ks = 0; ks < 8; ++ks)
    qa[ks] = load_a_frag(qrow, P_, lane, ks * 32);

  for (int i = 0; i < 8; ++i) {
    int ct = wave * 8 + i;                       // 64 column tiles over N
    const __bf16* krow = kmat + ((size_t)b * N_ + ct * 16) * P_;
    v8f acc = {0.f, 0.f, 0.f, 0.f, 0.f, 0.f, 0.f, 0.f};
#pragma unroll
    for (int ks = 0; ks < 8; ++ks) {
      v16bf bf = load_b_frag(krow, P_, lane, ks * 32);
      acc = __builtin_amdgcn_wmma_f32_16x16x32_bf16(false, qa[ks], false, bf,
                                                    (short)0, acc, false, false);
    }
#pragma unroll
    for (int j = 0; j < 8; ++j)
      S[mrow + j][ct * 16 + ncol] = acc[j];
  }
  __syncthreads();

  // ---- Phase B: softmax rows, temper = sqrt(256) = 16 ----
  for (int rr = 0; rr < 2; ++rr) {
    int r = wave * 2 + rr;
    float mx = -3.4e38f;
    for (int i = lane; i < N_; i += 32) mx = fmaxf(mx, S[r][i]);
#pragma unroll
    for (int off = 16; off; off >>= 1) mx = fmaxf(mx, __shfl_xor(mx, off, 32));
    float sum = 0.f;
    for (int i = lane; i < N_; i += 32) {
      float e = __expf((S[r][i] - mx) * 0.0625f);
      S[r][i] = e;
      sum += e;
    }
#pragma unroll
    for (int off = 16; off; off >>= 1) sum += __shfl_xor(sum, off, 32);
    float inv = 1.0f / sum;
    for (int i = lane; i < N_; i += 32) Pm[r][i] = (__bf16)(S[r][i] * inv);
  }
  __syncthreads();

  // ---- Phase C: out_tile = P @ x ----
  // One LDS A-fragment per K-step feeds TWO accumulators (the wave's two
  // D-column tiles) -> halves LDS read traffic, 2 WMMAs per loop body.
  {
    int ct0 = wave * 2;
    int ct1 = wave * 2 + 1;
    const __bf16* brow0 = xT + ((size_t)b * D_ + ct0 * 16) * N_;
    const __bf16* brow1 = xT + ((size_t)b * D_ + ct1 * 16) * N_;
    v8f acc0 = {0.f, 0.f, 0.f, 0.f, 0.f, 0.f, 0.f, 0.f};
    v8f acc1 = acc0;
    for (int ks = 0; ks < 32; ++ks) {
      v16bf afr = load_a_frag(&Pm[0][0], PPAD, lane, ks * 32);
      v16bf bf0 = load_b_frag(brow0, N_, lane, ks * 32);
      v16bf bf1 = load_b_frag(brow1, N_, lane, ks * 32);
      acc0 = __builtin_amdgcn_wmma_f32_16x16x32_bf16(false, afr, false, bf0,
                                                     (short)0, acc0, false, false);
      acc1 = __builtin_amdgcn_wmma_f32_16x16x32_bf16(false, afr, false, bf1,
                                                     (short)0, acc1, false, false);
    }
#pragma unroll
    for (int j = 0; j < 8; ++j) {
      outb[mrow + j][ct0 * 16 + ncol] = acc0[j];
      outb[mrow + j][ct1 * 16 + ncol] = acc1[j];
    }
  }
  __syncthreads();

  // ---- Phase D: logits = out_tile @ w4 ----
  for (int rr = 0; rr < 2; ++rr) {
    int r = wave * 2 + rr;
    float dot = 0.f;
    for (int d = lane; d < D_; d += 32) dot += outb[r][d] * w4[d];
#pragma unroll
    for (int off = 16; off; off >>= 1) dot += __shfl_xor(dot, off, 32);
    if (lane == 0) logits[(size_t)b * N_ + row0 + r] = dot;
  }
}

// ---------------------------------------------------------------------------
// Kernel 5: a = softmax_n(logits); sentence[b,d] = sum_n x[b,n,d]*a[n]
// ---------------------------------------------------------------------------
__global__ void __launch_bounds__(256) k_final(const float* __restrict__ x,
                                               const float* __restrict__ logits,
                                               float* __restrict__ out) {
  __shared__ float a[N_];
  __shared__ float red[256];
  int b = blockIdx.x;
  int tid = threadIdx.x;

  float mx = -3.4e38f;
  for (int i = tid; i < N_; i += 256) {
    float v = logits[(size_t)b * N_ + i];
    a[i] = v;
    mx = fmaxf(mx, v);
  }
  red[tid] = mx; __syncthreads();
  for (int s = 128; s; s >>= 1) { if (tid < s) red[tid] = fmaxf(red[tid], red[tid + s]); __syncthreads(); }
  mx = red[0]; __syncthreads();

  float sum = 0.f;
  for (int i = tid; i < N_; i += 256) {
    float e = __expf(a[i] - mx);
    a[i] = e;
    sum += e;
  }
  red[tid] = sum; __syncthreads();
  for (int s = 128; s; s >>= 1) { if (tid < s) red[tid] += red[tid + s]; __syncthreads(); }
  float inv = 1.0f / red[0];
  __syncthreads();

  int d = tid;                                   // D_ == blockDim.x == 256
  float acc = 0.f;
  const float* xb = x + (size_t)b * N_ * D_;
  for (int n = 0; n < N_; ++n) acc += xb[(size_t)n * D_ + d] * a[n];
  out[(size_t)b * D_ + d] = acc * inv;
}

// ---------------------------------------------------------------------------
extern "C" void kernel_launch(void* const* d_in, const int* in_sizes, int n_in,
                              void* d_out, int out_size, void* d_ws, size_t ws_size,
                              hipStream_t stream) {
  const float* x  = (const float*)d_in[0];
  const float* w1 = (const float*)d_in[1];
  const float* w2 = (const float*)d_in[2];
  const float* w4 = (const float*)d_in[3];
  float* out = (float*)d_out;

  char* ws = (char*)d_ws;
  size_t off = 0;
  auto alloc = [&](size_t bytes) {
    char* p = ws + off;
    off += (bytes + 255) & ~(size_t)255;
    return p;
  };
  __bf16* xbf    = (__bf16*)alloc((size_t)B_ * N_ * D_ * 2);
  __bf16* xT     = (__bf16*)alloc((size_t)B_ * D_ * N_ * 2);
  __bf16* w1T    = (__bf16*)alloc((size_t)P_ * D_ * 2);
  __bf16* w2T    = (__bf16*)alloc((size_t)P_ * D_ * 2);
  __bf16* qb     = (__bf16*)alloc((size_t)B_ * N_ * P_ * 2);
  __bf16* kb     = (__bf16*)alloc((size_t)B_ * N_ * P_ * 2);
  float*  logits = (float*)alloc((size_t)B_ * N_ * 4);

  k_prep_x<<<dim3(D_ / 16, N_ / 16, B_), dim3(16, 16), 0, stream>>>(x, xbf, xT);
  k_prep_w<<<dim3(P_ / 16, D_ / 16, 2), dim3(16, 16), 0, stream>>>(w1, w2, w1T, w2T);
  k_qk<<<dim3((B_ * N_ / 16) / 8), 256, 0, stream>>>(xbf, w1T, w2T, qb, kb);
  k_attn<<<dim3(N_ / 16, B_), 256, 0, stream>>>(qb, kb, xT, w4, logits);
  k_final<<<dim3(B_), 256, 0, stream>>>(x, logits, out);
}
